// STRNN_84155589198709
// MI455X (gfx1250) — matrix operand order repaired and checked by reading, not compiled, not run on hardware
//
#include <hip/hip_runtime.h>
#include <hip/hip_bf16.h>

#define L_LEN   131072
#define HDIM    256
#define NBLK    256                         // gather-reduce blocks
#define WPB     8                           // wave32s per block
#define ROWS_PER_WAVE (L_LEN / (NBLK * WPB))  // 64

typedef __attribute__((ext_vector_type(16))) _Float16 v16h;
typedef __attribute__((ext_vector_type(8)))  float    v8f;

// ---------------------------------------------------------------------------
// Kernel 1: bandwidth-bound gather + 4-way weighted reduction of embeddings.
// Each wave processes whole rows; lane holds 8 contiguous H-elements.
// ---------------------------------------------------------------------------
__global__ __launch_bounds__(256) void strnn_gather_reduce(
    const float* __restrict__ td_u, const float* __restrict__ td_l,
    const float* __restrict__ ld_u, const float* __restrict__ ld_l,
    const int*   __restrict__ loc,  const float* __restrict__ locw,
    float* __restrict__ partials)
{
    __shared__ float wred[WPB][4 * HDIM];

    const int tid  = threadIdx.x;
    const int lane = tid & 31;
    const int w    = tid >> 5;
    // wave-uniform row base -> scalar (s_load) accesses for the per-row data
    const int gw = __builtin_amdgcn_readfirstlane((int)(blockIdx.x * WPB + w));
    const int l0 = gw * ROWS_PER_WAVE;

    float acc[4][8];
#pragma unroll
    for (int j = 0; j < 4; ++j)
#pragma unroll
        for (int i = 0; i < 8; ++i) acc[j][i] = 0.f;

    for (int it = 0; it < ROWS_PER_WAVE; ++it) {
        const int l = l0 + it;

        // prefetch a future gather row (stays inside this wave's row range)
        {
            const int lp = l0 + ((it + 8) & (ROWS_PER_WAVE - 1));
            const int rp = loc[lp];
            __builtin_prefetch(locw + (size_t)rp * HDIM + lane * 8, 0, 0);
        }

        float tu = fminf(fmaxf(td_u[l], 0.f), 1440.f);
        float tb = fminf(fmaxf(td_l[l], 0.f), 1440.f);
        float lu = fminf(fmaxf(ld_u[l], 0.f), 40.f);
        float lb = fminf(fmaxf(ld_l[l], 0.f), 40.f);
        const float invT = 1.f / (tu + tb);
        const float invL = 1.f / (lu + lb);
        const float g = tu * invT, d = tb * invT;   // gamma, delta
        const float a = lu * invL, b = lb * invL;   // alpha, beta
        const float wj[4] = { a * g, a * d, b * g, b * d };

        const int r = loc[l];
        const float4* ep =
            (const float4*)(locw + (size_t)r * HDIM) + lane * 2;
        const float4 e0 = ep[0];
        const float4 e1 = ep[1];

#pragma unroll
        for (int j = 0; j < 4; ++j) {
            acc[j][0] += wj[j] * e0.x;  acc[j][1] += wj[j] * e0.y;
            acc[j][2] += wj[j] * e0.z;  acc[j][3] += wj[j] * e0.w;
            acc[j][4] += wj[j] * e1.x;  acc[j][5] += wj[j] * e1.y;
            acc[j][6] += wj[j] * e1.z;  acc[j][7] += wj[j] * e1.w;
        }
    }

#pragma unroll
    for (int j = 0; j < 4; ++j)
#pragma unroll
        for (int i = 0; i < 8; ++i)
            wred[w][j * HDIM + lane * 8 + i] = acc[j][i];
    __syncthreads();

    for (int idx = tid; idx < 4 * HDIM; idx += 256) {
        float s = 0.f;
#pragma unroll
        for (int ww = 0; ww < WPB; ++ww) s += wred[ww][idx];
        partials[(size_t)blockIdx.x * (4 * HDIM) + idx] = s;
    }
}

// ---------------------------------------------------------------------------
// Kernel 2: reduce per-block partials -> v[4][256]
// ---------------------------------------------------------------------------
__global__ __launch_bounds__(256) void strnn_partial_sum(
    const float* __restrict__ partials, float* __restrict__ vbuf)
{
    const int k = blockIdx.x * 256 + threadIdx.x;   // 0..1023
    float s = 0.f;
    for (int b = 0; b < NBLK; ++b) s += partials[(size_t)b * (4 * HDIM) + k];
    vbuf[k] = s;
}

// ---------------------------------------------------------------------------
// WMMA helper: one 16-row M-tile of Y = W(256x256,f32) @ X(256x16,f16 col-major
// in LDS), accumulated in f32 via v_wmma_f32_16x16x32_f16.
// A-frag (16-bit 16x32): lane m = lane&15; half = lane>>4 selects K quad.
// B-frag (32x16): lane n = lane&15; half selects K quad (same K order as A).
// ---------------------------------------------------------------------------
__device__ inline v8f wmma_gemm_row_tile(const float* __restrict__ W,
                                         const _Float16* Xcm,
                                         int mtile, int lane)
{
    const int m    = mtile * 16 + (lane & 15);
    const int half = lane >> 4;
    const int n    = lane & 15;
    v8f acc = {};
#pragma unroll
    for (int s = 0; s < 8; ++s) {
        const int kb = s * 32;
        v16h a, b;
        const float* wr = W + (size_t)m * HDIM + kb + half * 8;
#pragma unroll
        for (int i = 0; i < 8; ++i) {
            a[i]     = (_Float16)wr[i];
            a[8 + i] = (_Float16)wr[16 + i];
        }
        const _Float16* xc = Xcm + n * HDIM + kb + half * 8;
#pragma unroll
        for (int i = 0; i < 8; ++i) {
            b[i]     = xc[i];
            b[8 + i] = xc[16 + i];
        }
        acc = __builtin_amdgcn_wmma_f32_16x16x32_f16(
            false, a, false, b, (short)0, acc, false, false);
    }
    return acc;
}

// ---------------------------------------------------------------------------
// Kernel 3: tiny WMMA head.
//   X cols: [v1 v2 v3 v4 hx 0..]
//   stage1: A1=W_tu@X, A2=W_tl@X, A3=W_ih@X ; p=A1c0+A2c1, q=A1c2+A2c3,
//           usr=A3c4
//   stage2: loc = (W_su@P)c0 + (W_sl@P)c1, P = [p q 0..]
//   out = sigmoid(loc + usr)
// ---------------------------------------------------------------------------
__global__ __launch_bounds__(512) void strnn_head(
    const float* __restrict__ vbuf, const float* __restrict__ hx,
    const float* __restrict__ W_tu, const float* __restrict__ W_tl,
    const float* __restrict__ W_ih, const float* __restrict__ W_su,
    const float* __restrict__ W_sl, float* __restrict__ out)
{
    __shared__ _Float16 Xcm[16 * HDIM];
    __shared__ _Float16 Pcm[16 * HDIM];
    __shared__ float S1[HDIM][2];
    __shared__ float S2[HDIM][2];
    __shared__ float U[HDIM];
    __shared__ float R0[HDIM];
    __shared__ float R1[HDIM];

    const int tid  = threadIdx.x;
    const int lane = tid & 31;
    const int w    = tid >> 5;        // 0..15 == M-tile id

    for (int idx = tid; idx < 16 * HDIM; idx += 512) {
        const int col = idx >> 8;
        const int k   = idx & 255;
        float v = 0.f;
        if (col < 4)       v = vbuf[col * HDIM + k];
        else if (col == 4) v = hx[k];
        Xcm[idx] = (_Float16)v;
    }
    __syncthreads();

    v8f d1 = wmma_gemm_row_tile(W_tu, Xcm, w, lane);
    v8f d2 = wmma_gemm_row_tile(W_tl, Xcm, w, lane);
    v8f d3 = wmma_gemm_row_tile(W_ih, Xcm, w, lane);

    const int n    = lane & 15;
    const int half = lane >> 4;
#pragma unroll
    for (int r = 0; r < 8; ++r) {
        const int row = w * 16 + half * 8 + r;   // C/D layout: vgpr r -> M
        if (n == 0) S1[row][0] = d1[r];
        if (n == 2) S1[row][1] = d1[r];
        if (n == 1) S2[row][0] = d2[r];
        if (n == 3) S2[row][1] = d2[r];
        if (n == 4) U[row]     = d3[r];
    }
    __syncthreads();

    for (int idx = tid; idx < 16 * HDIM; idx += 512) {
        const int col = idx >> 8;
        const int k   = idx & 255;
        float v = 0.f;
        if (col == 0)      v = S1[k][0] + S2[k][0];   // p
        else if (col == 1) v = S1[k][1] + S2[k][1];   // q
        Pcm[idx] = (_Float16)v;
    }
    __syncthreads();

    v8f d4 = wmma_gemm_row_tile(W_su, Pcm, w, lane);
    v8f d5 = wmma_gemm_row_tile(W_sl, Pcm, w, lane);
#pragma unroll
    for (int r = 0; r < 8; ++r) {
        const int row = w * 16 + half * 8 + r;
        if (n == 0) R0[row] = d4[r];
        if (n == 1) R1[row] = d5[r];
    }
    __syncthreads();

    if (tid < HDIM) {
        const float z = R0[tid] + R1[tid] + U[tid];
        out[tid] = 1.f / (1.f + __expf(-z));
    }
}

// ---------------------------------------------------------------------------
extern "C" void kernel_launch(void* const* d_in, const int* in_sizes, int n_in,
                              void* d_out, int out_size, void* d_ws, size_t ws_size,
                              hipStream_t stream)
{
    const float* td_u = (const float*)d_in[0];
    const float* td_l = (const float*)d_in[1];
    const float* ld_u = (const float*)d_in[2];
    const float* ld_l = (const float*)d_in[3];
    const int*   loc  = (const int*)  d_in[4];
    const float* hx   = (const float*)d_in[5];
    const float* W_ih = (const float*)d_in[6];
    const float* W_tu = (const float*)d_in[7];
    const float* W_tl = (const float*)d_in[8];
    const float* W_su = (const float*)d_in[9];
    const float* W_sl = (const float*)d_in[10];
    const float* locw = (const float*)d_in[11];

    float* partials = (float*)d_ws;                       // NBLK * 1024 f32
    float* vbuf     = partials + (size_t)NBLK * 4 * HDIM; // 1024 f32

    strnn_gather_reduce<<<NBLK, 256, 0, stream>>>(
        td_u, td_l, ld_u, ld_l, loc, locw, partials);
    strnn_partial_sum<<<4, 256, 0, stream>>>(partials, vbuf);
    strnn_head<<<1, 512, 0, stream>>>(
        vbuf, hx, W_tu, W_tl, W_ih, W_su, W_sl, (float*)d_out);
}
